// LSM_Branch_23184233464093
// MI455X (gfx1250) — matrix operand order repaired and checked by reading, not compiled, not run on hardware
//
#include <hip/hip_runtime.h>

typedef __attribute__((ext_vector_type(16))) __bf16        v16bf;
typedef __attribute__((ext_vector_type(8)))  float         v8f;
typedef __attribute__((ext_vector_type(8)))  unsigned int  v8u;

#define BM   128   // block tile M
#define BNT  64    // block tile N
#define KC   32    // K chunk (one wmma K)
#define LST  40    // LDS row stride (ushorts) = 80B: 16B-aligned rows, bank-rotating
#define ASZ  (BM * LST)   // one A buffer (ushorts)
#define BSZ  (BNT * LST)  // one B buffer (ushorts)

__device__ __forceinline__ unsigned short f2bf(float f) {
    unsigned u = __float_as_uint(f);
    u += 0x7fffu + ((u >> 16) & 1u);      // round-to-nearest-even
    return (unsigned short)(u >> 16);
}

// ---- CDNA5 async global->LDS copy (ASYNCcnt-tracked, bypasses VGPRs) ----
// GVS mode: mem_addr = SGPR[base] + VGPR[voff]; LDS addr = per-lane VGPR (low 32
// bits of the generic pointer to __shared__ = LDS byte offset).
__device__ __forceinline__ void async_cp16(const void* gbase, unsigned voff,
                                           const unsigned short* ldsdst) {
    unsigned loff = (unsigned)(uintptr_t)ldsdst;
    asm volatile("global_load_async_to_lds_b128 %0, %1, %2"
                 :: "v"(loff), "v"(voff), "s"(gbase) : "memory");
}
__device__ __forceinline__ void async_wait0() {
    asm volatile("s_wait_asynccnt 0x0" ::: "memory");
}

// stage NR x 32 bf16 tile via async DMA: one b128 (8 bf16) per 16B chunk
template <int NR>
__device__ __forceinline__ void stage_bf16_async(
    const unsigned short* __restrict__ G, int ldg, int row0, int k0,
    unsigned short* lds, int tid)
{
    #pragma unroll
    for (int idx = 0; idx < NR * 4; idx += 256) {
        int r   = (idx + tid) >> 2;
        int seg = (idx + tid) & 3;
        unsigned voff = (((unsigned)(row0 + r) * (unsigned)ldg + (unsigned)k0) << 1)
                        + (unsigned)(seg << 4);
        async_cp16(G, voff, lds + r * LST + seg * 8);
    }
}

// stage 128 x 32 fp32 tile with convert (VGPR path; only used for x in GEMM1)
__device__ __forceinline__ void stage_f32(
    const float* __restrict__ G, int ldg, int row0, int k0,
    unsigned short* lds, int tid)
{
    const int lrow = tid >> 3, lcol = (tid & 7) << 2;
    #pragma unroll
    for (int rr = 0; rr < BM; rr += 32) {
        int r = lrow + rr;
        float4 v = *reinterpret_cast<const float4*>(G + (size_t)(row0 + r) * ldg + k0 + lcol);
        unsigned short* d = lds + r * LST + lcol;
        d[0] = f2bf(v.x); d[1] = f2bf(v.y); d[2] = f2bf(v.z); d[3] = f2bf(v.w);
    }
}

__device__ __forceinline__ void stage_a(const float* G, int ldg, int m0, int k0,
                                        unsigned short* lds, int tid)
{ stage_f32(G, ldg, m0, k0, lds, tid); }
__device__ __forceinline__ void stage_a(const unsigned short* G, int ldg, int m0, int k0,
                                        unsigned short* lds, int tid)
{ stage_bf16_async<BM>(G, ldg, m0, k0, lds, tid); }

// Block GEMM core, double-buffered async pipeline:
// acc[2][2] += A[m0:m0+128, :K] * B[n0:n0+64, :K]^T  (bf16 wmma, f32 acc)
template <typename AT>
__device__ __forceinline__ void gemm_core(
    const AT* __restrict__ A, int lda,
    const unsigned short* __restrict__ B, int ldb,
    int K, int m0, int n0,
    unsigned short* a_s, unsigned short* b_s,   // 2 buffers each
    v8f acc[2][2])
{
    const int tid  = threadIdx.x;
    const int lane = tid & 31;
    const int half = lane >> 4;
    const int ml   = lane & 15;
    const int wave = tid >> 5;
    const int wm   = wave >> 1;     // 0..3
    const int wn   = wave & 1;      // 0..1

    if (K > 0) {                                   // prologue: fill buffer 0
        stage_a(A, lda, m0, 0, a_s, tid);
        stage_bf16_async<BNT>(B, ldb, n0, 0, b_s, tid);
    }
    int cur = 0;
    for (int k0 = 0; k0 < K; k0 += KC) {
        async_wait0();                 // my async copies for buf[cur] landed
        __syncthreads();               // everyone's copies landed / reads done
        const int nxt = cur ^ 1;
        if (k0 + KC < K) {             // prefetch next chunk during compute
            stage_a(A, lda, m0, k0 + KC, a_s + nxt * ASZ, tid);
            stage_bf16_async<BNT>(B, ldb, n0, k0 + KC, b_s + nxt * BSZ, tid);
        }
        const unsigned short* as = a_s + cur * ASZ;
        const unsigned short* bs = b_s + cur * BSZ;

        v8u au[2], bu[2];
        #pragma unroll
        for (int i = 0; i < 2; ++i) {              // A 16x32 bf16 layout (ISA 7.12.2)
            const unsigned short* ap = &as[(wm * 32 + i * 16 + ml) * LST];
            #pragma unroll
            for (int v = 0; v < 4; ++v)
                au[i][v]     = *reinterpret_cast<const unsigned int*>(ap + half * 8 + 2 * v);
            #pragma unroll
            for (int v = 0; v < 4; ++v)
                au[i][4 + v] = *reinterpret_cast<const unsigned int*>(ap + 16 + half * 8 + 2 * v);
        }
        #pragma unroll
        for (int j = 0; j < 2; ++j) {              // B 32x16 bf16 layout
            const unsigned short* bp = &bs[(wn * 32 + j * 16 + ml) * LST];
            #pragma unroll
            for (int v = 0; v < 8; ++v)
                bu[j][v] = *reinterpret_cast<const unsigned int*>(bp + half * 8 + 2 * v + (v >> 2) * 8);
        }
        #pragma unroll
        for (int i = 0; i < 2; ++i)
            #pragma unroll
            for (int j = 0; j < 2; ++j)
                acc[i][j] = __builtin_amdgcn_wmma_f32_16x16x32_bf16(
                    false, __builtin_bit_cast(v16bf, au[i]),
                    false, __builtin_bit_cast(v16bf, bu[j]),
                    (short)0, acc[i][j], false, false);
        cur = nxt;
    }
}

// ---------------- kernels ----------------

__global__ void k_zero(float* p, size_t n) {
    for (size_t i = (size_t)blockIdx.x * blockDim.x + threadIdx.x; i < n;
         i += (size_t)gridDim.x * blockDim.x) p[i] = 0.f;
}

__global__ void k_cvt(const float* __restrict__ src, unsigned short* __restrict__ dst, int n) {
    int i = blockIdx.x * 256 + threadIdx.x;
    if (i < n) dst[i] = f2bf(src[i]);
}

// curr = x @ W1^T + b1   (A fp32->bf16 on the fly)
__global__ void k_gemm1(const float* __restrict__ X, const unsigned short* __restrict__ W1b,
                        const float* __restrict__ b1, float* __restrict__ curr)
{
    __shared__ unsigned short a_s[2 * ASZ];
    __shared__ unsigned short b_s[2 * BSZ];
    const int n0 = blockIdx.x * BNT, m0 = blockIdx.y * BM;
    v8f acc[2][2] = {};
    gemm_core(X, 512, W1b, 512, 512, m0, n0, a_s, b_s, acc);
    const int tid = threadIdx.x, lane = tid & 31, half = lane >> 4, nl = lane & 15;
    const int wave = tid >> 5, wm = wave >> 1, wn = wave & 1;
    #pragma unroll
    for (int i = 0; i < 2; ++i)
      #pragma unroll
      for (int j = 0; j < 2; ++j)
        #pragma unroll
        for (int r = 0; r < 8; ++r) {
            int row = m0 + wm * 32 + i * 16 + half * 8 + r;
            int col = n0 + wn * 32 + j * 16 + nl;
            curr[(size_t)row * 1024 + col] = acc[i][j][r] + b1[col];
        }
}

// one scan step: inp = curr_t + spk_prev @ Wrec^T + brec; PLIF charge/fire/reset
__global__ void k_rec(const unsigned short* __restrict__ SpkPrev, int Kdim,
                      const unsigned short* __restrict__ Wrecb,
                      const float* __restrict__ brec,
                      const float* __restrict__ curr_t,
                      const float* __restrict__ w_lif,
                      float* __restrict__ vmem,
                      unsigned short* __restrict__ SpkOut)
{
    __shared__ unsigned short a_s[2 * ASZ];
    __shared__ unsigned short b_s[2 * BSZ];
    const int n0 = blockIdx.x * BNT, m0 = blockIdx.y * BM;
    v8f acc[2][2] = {};
    gemm_core(SpkPrev, 1024, Wrecb, 1024, Kdim, m0, n0, a_s, b_s, acc);
    const float decay = 1.0f / (1.0f + __expf(-w_lif[0]));
    const int tid = threadIdx.x, lane = tid & 31, half = lane >> 4, nl = lane & 15;
    const int wave = tid >> 5, wm = wave >> 1, wn = wave & 1;
    #pragma unroll
    for (int i = 0; i < 2; ++i)
      #pragma unroll
      for (int j = 0; j < 2; ++j)
        #pragma unroll
        for (int r = 0; r < 8; ++r) {
            int row = m0 + wm * 32 + i * 16 + half * 8 + r;
            int col = n0 + wn * 32 + j * 16 + nl;
            size_t idx = (size_t)row * 1024 + col;
            float inp = acc[i][j][r] + brec[col] + curr_t[idx];
            float vv  = vmem[idx];
            vv = fmaf(inp - vv, decay, vv);                 // charge
            float s = (vv >= 1.0f) ? 1.0f : 0.0f;           // fire (V_TH = 1)
            vmem[idx]  = vv * (1.0f - s);                   // hard reset
            SpkOut[idx] = f2bf(s);                          // {0,1} exact in bf16
        }
}

// out = spks @ W2^T + b2  (pre-BN, fp32)
__global__ void k_gemm2(const unsigned short* __restrict__ Spk,
                        const unsigned short* __restrict__ W2b,
                        const float* __restrict__ b2, float* __restrict__ out)
{
    __shared__ unsigned short a_s[2 * ASZ];
    __shared__ unsigned short b_s[2 * BSZ];
    const int n0 = blockIdx.x * BNT, m0 = blockIdx.y * BM;
    v8f acc[2][2] = {};
    gemm_core(Spk, 1024, W2b, 1024, 1024, m0, n0, a_s, b_s, acc);
    const int tid = threadIdx.x, lane = tid & 31, half = lane >> 4, nl = lane & 15;
    const int wave = tid >> 5, wm = wave >> 1, wn = wave & 1;
    #pragma unroll
    for (int i = 0; i < 2; ++i)
      #pragma unroll
      for (int j = 0; j < 2; ++j)
        #pragma unroll
        for (int r = 0; r < 8; ++r) {
            int row = m0 + wm * 32 + i * 16 + half * 8 + r;
            int col = n0 + wn * 32 + j * 16 + nl;
            out[(size_t)row * 512 + col] = acc[i][j][r] + b2[col];
        }
}

// BN stats pass 1: each block reduces 64 rows; thread t owns channels t and t+256
__global__ void k_bnstat(const float* __restrict__ out,
                         float* __restrict__ psum, float* __restrict__ psq)
{
    const int tid = threadIdx.x;
    const size_t base = (size_t)blockIdx.x * 64 * 512;
    float s0 = 0.f, q0 = 0.f, s1 = 0.f, q1 = 0.f;
    for (int k = 0; k < 128; k += 2) {
        float x0 = out[base + tid + (size_t)256 * k];        // channel tid
        float x1 = out[base + tid + (size_t)256 * (k + 1)];  // channel tid+256
        s0 += x0; q0 += x0 * x0;
        s1 += x1; q1 += x1 * x1;
    }
    size_t b = (size_t)blockIdx.x * 512;
    psum[b + tid] = s0;        psum[b + tid + 256] = s1;
    psq [b + tid] = q0;        psq [b + tid + 256] = q1;
}

// BN stats pass 2: fixed-order reduction over 1024 block partials (deterministic)
__global__ void k_bnfinal(const float* __restrict__ psum, const float* __restrict__ psq,
                          float* __restrict__ sums)
{
    int c = blockIdx.x * 256 + threadIdx.x;   // 0..511
    if (c >= 512) return;
    float s = 0.f, q = 0.f;
    for (int b = 0; b < 1024; ++b) { s += psum[(size_t)b * 512 + c]; q += psq[(size_t)b * 512 + c]; }
    sums[c] = s; sums[512 + c] = q;
}

__global__ void k_bnnorm(float* __restrict__ out, const float* __restrict__ sums,
                         const float* __restrict__ gamma, const float* __restrict__ beta,
                         size_t n)
{
    const float invM = 1.0f / 65536.0f;
    for (size_t idx = (size_t)blockIdx.x * 256 + threadIdx.x; idx < n;
         idx += (size_t)gridDim.x * 256) {
        int c = (int)(idx & 511);
        float mean = sums[c] * invM;
        float var  = sums[512 + c] * invM - mean * mean;    // biased, matches reference
        float x = out[idx];
        out[idx] = (x - mean) * rsqrtf(var + 1e-5f) * gamma[c] + beta[c];
    }
}

extern "C" void kernel_launch(void* const* d_in, const int* in_sizes, int n_in,
                              void* d_out, int out_size, void* d_ws, size_t ws_size,
                              hipStream_t stream)
{
    const float* x     = (const float*)d_in[0];
    const float* W1    = (const float*)d_in[1];
    const float* b1    = (const float*)d_in[2];
    const float* Wrec  = (const float*)d_in[3];
    const float* brec  = (const float*)d_in[4];
    const float* wlif  = (const float*)d_in[5];
    const float* W2    = (const float*)d_in[6];
    const float* b2    = (const float*)d_in[7];
    const float* gamma = (const float*)d_in[8];
    const float* beta  = (const float*)d_in[9];
    float* out = (float*)d_out;

    const int T = 32, BNr = 2048, C = 512, H = 1024, O = 512;
    const size_t MR = (size_t)T * BNr;            // 65536 total rows

    char* ws = (char*)d_ws;
    float*          curr   = (float*)ws;          ws += MR * H * 4;            // 268 MB
    unsigned short* spikes = (unsigned short*)ws; ws += MR * H * 2;            // 134 MB (bf16)
    float*          vmem   = (float*)ws;          ws += (size_t)BNr * H * 4;   // 8 MB
    unsigned short* W1b    = (unsigned short*)ws; ws += (size_t)H * C * 2;
    unsigned short* Wrecb  = (unsigned short*)ws; ws += (size_t)H * H * 2;
    unsigned short* W2b    = (unsigned short*)ws; ws += (size_t)O * H * 2;
    float*          psum   = (float*)ws;          ws += (size_t)1024 * 512 * 4;
    float*          psq    = (float*)ws;          ws += (size_t)1024 * 512 * 4;
    float*          sums   = (float*)ws;          ws += 1024 * 4;

    // membrane potential starts at 0 every launch (harness does not re-poison)
    k_zero<<<512, 256, 0, stream>>>(vmem, (size_t)BNr * H);

    // weights -> bf16 once
    k_cvt<<<(H * C + 255) / 256, 256, 0, stream>>>(W1,   W1b,   H * C);
    k_cvt<<<(H * H + 255) / 256, 256, 0, stream>>>(Wrec, Wrecb, H * H);
    k_cvt<<<(O * H + 255) / 256, 256, 0, stream>>>(W2,   W2b,   O * H);

    // Phase 1: input projection, all timesteps at once
    k_gemm1<<<dim3(H / BNT, MR / BM), 256, 0, stream>>>(x, W1b, b1, curr);

    // Phase 2: sequential recurrent scan (t=0 has zero spikes -> K=0 skips GEMM)
    for (int t = 0; t < T; ++t) {
        const unsigned short* spk_prev =
            (t > 0) ? spikes + (size_t)(t - 1) * BNr * H : spikes;
        k_rec<<<dim3(H / BNT, BNr / BM), 256, 0, stream>>>(
            spk_prev, (t > 0) ? H : 0, Wrecb, brec,
            curr + (size_t)t * BNr * H, wlif,
            vmem, spikes + (size_t)t * BNr * H);
    }

    // Phase 3: readout projection
    k_gemm2<<<dim3(O / BNT, MR / BM), 256, 0, stream>>>(spikes, W2b, b2, out);

    // Phase 4: BatchNorm (deterministic 2-pass stats, then in-place normalize)
    k_bnstat<<<(int)(MR / 64), 256, 0, stream>>>(out, psum, psq);
    k_bnfinal<<<2, 256, 0, stream>>>(psum, psq, sums);
    k_bnnorm<<<8192, 256, 0, stream>>>(out, sums, gamma, beta, MR * (size_t)O);
}
